// MultiHeadAttention_60181081751875
// MI455X (gfx1250) — compile-verified, matrix operand order checked by reading, not run on hardware
//
#include <hip/hip_runtime.h>

typedef __attribute__((ext_vector_type(16))) _Float16 v16h;
typedef __attribute__((ext_vector_type(8)))  _Float16 v8h;
typedef __attribute__((ext_vector_type(4)))  _Float16 v4h;
typedef __attribute__((ext_vector_type(8)))  float    v8f;

// Problem constants: N nodes, B batch, E embed, H heads, ED edge dim
static constexpr int NN = 1024;
static constexpr int BB = 8;
static constexpr int EE = 256;
static constexpr int NH = 8;
static constexpr int ED = 16;
static constexpr int HD = 32;      // E / H
static constexpr int PPAD = 1032;  // padded prob-row stride (f16) to spread LDS banks

// K index inside a 16x16x32 f16 WMMA fragment (ISA 7.12.2):
// halves 0..7 <-> K = lanehalf*8 + h ; halves 8..15 <-> K = 16 + lanehalf*8 + (h-8)
__device__ __forceinline__ int kmap16(int kh, int h) {
  return (h < 8) ? (kh * 8 + h) : (16 + kh * 8 + (h - 8));
}

// ---------------------------------------------------------------------------
// Kernel 1: vproj[r][o] = sum_k value[r][k] * in_w[o][k] + in_b[o]
// ---------------------------------------------------------------------------
__global__ __launch_bounds__(256)
void vproj_kernel(const float* __restrict__ A, const float* __restrict__ Bm,
                  const float* __restrict__ bias, float* __restrict__ out) {
  const int wave = threadIdx.x >> 5;
  const int lane = threadIdx.x & 31;
  const int tile = blockIdx.x * 8 + wave;
  const int trow = tile >> 4;
  const int tcol = tile & 15;
  const int m  = lane & 15;
  const int kh = lane >> 4;

  const float bv = bias[tcol * 16 + m];
  v8f acc;
  #pragma unroll
  for (int r = 0; r < 8; ++r) acc[r] = bv;

  const float* arow = A  + (size_t)(trow * 16 + m) * EE;
  const float* brow = Bm + (size_t)(tcol * 16 + m) * EE;

  #pragma unroll
  for (int kb = 0; kb < EE; kb += 32) {
    v16h a, b;
    #pragma unroll
    for (int h = 0; h < 16; ++h) {
      const int k = kb + kmap16(kh, h);
      a[h] = (_Float16)arow[k];
      b[h] = (_Float16)brow[k];
    }
    acc = __builtin_amdgcn_wmma_f32_16x16x32_f16(false, a, false, b,
                                                 (short)0, acc, false, false);
  }
  #pragma unroll
  for (int r = 0; r < 8; ++r)
    out[(size_t)(trow * 16 + kh * 8 + r) * EE + tcol * 16 + m] = acc[r];
}

// ---------------------------------------------------------------------------
// Kernel 2: fused logits(WMMA) + mask + softmax(wave-local) + attn@v (WMMA).
//   One 256-thread WG per (bh, 16-row tile): 64*64 = 4096 WGs.
//   adj read exactly once (512 MB total); v read from L2 via coalesced staging.
// ---------------------------------------------------------------------------
__global__ __launch_bounds__(256)
void attn_fused_kernel(const float* __restrict__ adj,
                       const unsigned char* __restrict__ amask,
                       const float* __restrict__ gamma,
                       const float* __restrict__ beta,
                       const float* __restrict__ edge_w,
                       const float* __restrict__ edge_b,
                       const float* __restrict__ vp,     // [N*B][E]
                       float* __restrict__ ctx) {        // [64][N][HD]
  // 64 KB region time-shared: phase A = f16 adj slabs [row][e][q],
  // phase B = f16 value tile [j][d]. Separated by a __syncthreads().
  __shared__ __align__(16) unsigned char sBigRaw[16 * ED * 128 * 2];
  _Float16 (*sAdj)[ED][128] = (_Float16(*)[ED][128])sBigRaw;   // [16][16][128]
  _Float16 (*sV)[HD]        = (_Float16(*)[HD])sBigRaw;        // [1024][32]

  __shared__ float    sWef[NH][ED];
  __shared__ float    sBef[NH];
  __shared__ _Float16 sP[16][PPAD];      // softmax probs (padded rows)
  __shared__ float    sPart[8][16][16];  // per-wave partial attn@v tiles

  const int t   = threadIdx.x;
  const int bh  = blockIdx.x >> 6;
  const int it0 = (blockIdx.x & 63) * 16;
  const int b   = bh >> 3;
  const int hk  = bh & 7;

  // weff[h][e] = edge_w[h][e]*gamma[e]; beff[h] = edge_b[h] + sum_e edge_w*beta
  if (t < NH * ED) sWef[t >> 4][t & 15] = edge_w[t] * gamma[t & 15];
  if (t < NH) {
    float s = edge_b[t];
    #pragma unroll
    for (int e = 0; e < ED; ++e) s += edge_w[t * ED + e] * beta[e];
    sBef[t] = s;
  }
  __syncthreads();

  const int lane = t & 31;
  const int wv   = t >> 5;
  const int n  = lane & 15;
  const int kh = lane >> 4;

  // ---- Phase A1: stage this wave's two adj slabs into LDS as f16 ----------
  {
    const int i  = t >> 4;                 // wave w owns rows 2w, 2w+1
    const int qg = t & 15;
    const int gi = it0 + i;
    const int p  = hk * 128 + (gi >> 3);
    const int qb = (gi & 7) * 128;
    const float* abase = adj + (((size_t)b * ED) * NN + p) * NN + qb + qg * 8;
    #pragma unroll
    for (int e = 0; e < ED; ++e) {
      const float4* s4 = (const float4*)(abase + (size_t)e * NN * NN);
      const float4 x0 = s4[0], x1 = s4[1];
      if (e + 1 < ED)
        __builtin_prefetch(abase + (size_t)(e + 1) * NN * NN, 0, 0);
      v8h hv;
      hv[0] = (_Float16)x0.x; hv[1] = (_Float16)x0.y;
      hv[2] = (_Float16)x0.z; hv[3] = (_Float16)x0.w;
      hv[4] = (_Float16)x1.x; hv[5] = (_Float16)x1.y;
      hv[6] = (_Float16)x1.z; hv[7] = (_Float16)x1.w;
      *(v8h*)&sAdj[i][e][qg * 8] = hv;
    }
  }

  // Constant A fragment: weff padded to 16x32 (rows >=8 and K >=16 are zero).
  v16h afr;
  #pragma unroll
  for (int h = 0; h < 16; ++h)
    afr[h] = (h < 8 && n < 8) ? (_Float16)sWef[n][kh * 8 + h] : (_Float16)0.f;
  float cinit[8];
  #pragma unroll
  for (int r = 0; r < 8; ++r) cinit[r] = (kh == 0) ? sBef[r] : 0.f;

  // ---- Phase A2: logits via WMMA + wave-local masked softmax, per row -----
  #pragma unroll
  for (int rr = 0; rr < 2; ++rr) {
    const int ir  = wv * 2 + rr;
    const int gir = it0 + ir;

    v8f L[8];   // logits tiles: L[nt][r] = logit(hh = r + kh*8, q = nt*16 + n)
    #pragma unroll
    for (int nt = 0; nt < 8; ++nt) {
      v16h bf;
      #pragma unroll
      for (int h = 0; h < 8; ++h) bf[h] = sAdj[ir][kh * 8 + h][nt * 16 + n];
      #pragma unroll
      for (int h = 8; h < 16; ++h) bf[h] = (_Float16)0.f;
      v8f c;
      #pragma unroll
      for (int r = 0; r < 8; ++r) c[r] = cinit[r];
      L[nt] = __builtin_amdgcn_wmma_f32_16x16x32_f16(false, afr, false, bf,
                                                     (short)0, c, false, false);
    }

    // Valid logits all live in lanes 0..15 (kh==0): j = (nt*16+n)*8 + r.
    const unsigned char* mrow = amask + ((size_t)b * NN + gir) * NN;
    unsigned long long mk[8];
    float lmax = -1e30f;
    if (kh == 0) {
      #pragma unroll
      for (int nt = 0; nt < 8; ++nt) {
        mk[nt] = *(const unsigned long long*)(mrow + nt * 128 + n * 8);
        #pragma unroll
        for (int r = 0; r < 8; ++r)
          if ((mk[nt] >> (8 * r)) & 0xffull) lmax = fmaxf(lmax, L[nt][r]);
      }
    }
    #pragma unroll
    for (int off = 16; off > 0; off >>= 1)
      lmax = fmaxf(lmax, __shfl_xor(lmax, off, 32));

    float lsum = 0.f;
    if (kh == 0) {
      #pragma unroll
      for (int nt = 0; nt < 8; ++nt)
        #pragma unroll
        for (int r = 0; r < 8; ++r) {
          const float ev = ((mk[nt] >> (8 * r)) & 0xffull)
                               ? __expf(L[nt][r] - lmax) : 0.f;
          L[nt][r] = ev;
          lsum += ev;
        }
    }
    #pragma unroll
    for (int off = 16; off > 0; off >>= 1)
      lsum += __shfl_xor(lsum, off, 32);
    const float inv = 1.f / lsum;

    if (kh == 0) {
      #pragma unroll
      for (int nt = 0; nt < 8; ++nt) {
        v8h pv;
        #pragma unroll
        for (int r = 0; r < 8; ++r) pv[r] = (_Float16)(L[nt][r] * inv);
        *(v8h*)&sP[ir][(nt * 16 + n) * 8] = pv;
      }
    }
  }
  __syncthreads();   // probs visible to all waves; adj staging region now dead

  // ---- Phase B1: stage v2[bh] = vp[(j*B+b)*E + hk*32 + d] into LDS (f16) --
  {
    const int d0 = (t & 7) * 4;
    const int jb = t >> 3;               // 32 j-rows per pass
    for (int pass = 0; pass < 32; ++pass) {
      const int j = pass * 32 + jb;
      const float4 x = *(const float4*)&vp[((size_t)j * BB + b) * EE + hk * HD + d0];
      v4h hv;
      hv[0] = (_Float16)x.x; hv[1] = (_Float16)x.y;
      hv[2] = (_Float16)x.z; hv[3] = (_Float16)x.w;
      *(v4h*)&sV[j][d0] = hv;
    }
  }
  __syncthreads();

  // ---- Phase B2: ctx[16 x 32] = probs[16 x 1024] @ v2[1024 x 32] ----------
  // 8 waves = 4 K-chunks (256 each) x 2 d-halves (16 each).
  const int dhalf = wv & 1;
  const int jc    = wv >> 1;

  v8f o;
  #pragma unroll
  for (int r = 0; r < 8; ++r) o[r] = 0.f;

  for (int kb = jc * 256; kb < jc * 256 + 256; kb += 32) {
    const v8h alo = *(const v8h*)&sP[n][kb + kh * 8];
    const v8h ahi = *(const v8h*)&sP[n][kb + 16 + kh * 8];
    v16h a, bf;
    #pragma unroll
    for (int h = 0; h < 8; ++h) { a[h] = alo[h]; a[h + 8] = ahi[h]; }
    #pragma unroll
    for (int h = 0; h < 16; ++h)
      bf[h] = sV[kb + kmap16(kh, h)][dhalf * 16 + n];
    o = __builtin_amdgcn_wmma_f32_16x16x32_f16(false, a, false, bf,
                                               (short)0, o, false, false);
  }
  #pragma unroll
  for (int r = 0; r < 8; ++r) sPart[wv][kh * 8 + r][n] = o[r];
  __syncthreads();

  // Reduce the 4 K-chunk partials and store the context tile.
  {
    const int ii = t >> 4, dl = t & 15;
    #pragma unroll
    for (int dh = 0; dh < 2; ++dh) {
      const float s = sPart[0 + dh][ii][dl] + sPart[2 + dh][ii][dl] +
                      sPart[4 + dh][ii][dl] + sPart[6 + dh][ii][dl];
      ctx[((size_t)bh * NN + it0 + ii) * HD + dh * 16 + dl] = s;
    }
  }
}

// ---------------------------------------------------------------------------
// Kernel 3: out[r][o] = sum_e T[r][e] * out_w[o][e] + out_b[o]
//   r = n*B + b_i ;  T[r][e] = ctx[b_i*8 + (e>>5)][n][e&31]
// ---------------------------------------------------------------------------
__global__ __launch_bounds__(256)
void outproj_kernel(const float* __restrict__ ctx,
                    const float* __restrict__ Wout,
                    const float* __restrict__ bias, float* __restrict__ out) {
  const int wave = threadIdx.x >> 5;
  const int lane = threadIdx.x & 31;
  const int tile = blockIdx.x * 8 + wave;
  const int trow = tile >> 4;
  const int tcol = tile & 15;
  const int m  = lane & 15;
  const int kh = lane >> 4;

  const int r  = trow * 16 + m;
  const int ni = r >> 3;
  const int bi = r & 7;

  const float bv = bias[tcol * 16 + m];
  v8f acc;
  #pragma unroll
  for (int q = 0; q < 8; ++q) acc[q] = bv;

  const float* brow = Wout + (size_t)(tcol * 16 + m) * EE;

  #pragma unroll
  for (int kb = 0; kb < EE; kb += 32) {
    v16h a, b;
    #pragma unroll
    for (int h = 0; h < 16; ++h) {
      const int k = kb + kmap16(kh, h);
      a[h] = (_Float16)ctx[(((size_t)(bi * 8 + (k >> 5))) * NN + ni) * HD + (k & 31)];
      b[h] = (_Float16)brow[k];
    }
    acc = __builtin_amdgcn_wmma_f32_16x16x32_f16(false, a, false, b,
                                                 (short)0, acc, false, false);
  }
  #pragma unroll
  for (int q = 0; q < 8; ++q)
    out[(size_t)(trow * 16 + kh * 8 + q) * EE + tcol * 16 + m] = acc[q];
}

// ---------------------------------------------------------------------------
extern "C" void kernel_launch(void* const* d_in, const int* in_sizes, int n_in,
                              void* d_out, int out_size, void* d_ws, size_t ws_size,
                              hipStream_t stream) {
  (void)in_sizes; (void)n_in; (void)out_size; (void)ws_size;
  const float*         value  = (const float*)d_in[0];
  const float*         adj    = (const float*)d_in[1];
  const unsigned char* amask  = (const unsigned char*)d_in[2];
  const float*         gamma  = (const float*)d_in[3];
  const float*         beta   = (const float*)d_in[4];
  const float*         edge_w = (const float*)d_in[5];
  const float*         edge_b = (const float*)d_in[6];
  const float*         in_w   = (const float*)d_in[7];
  const float*         in_b   = (const float*)d_in[8];
  const float*         out_w  = (const float*)d_in[9];
  const float*         out_b  = (const float*)d_in[10];

  float* out = (float*)d_out;
  float* vp  = out;            // d_out doubles as vproj scratch; kernel 3 overwrites it
  float* ctx = (float*)d_ws;   // 64*1024*32 f32 = 8 MB attention context

  vproj_kernel<<<1024, 256, 0, stream>>>(value, in_w, in_b, vp);
  attn_fused_kernel<<<64 * 64, 256, 0, stream>>>(adj, amask, gamma, beta,
                                                 edge_w, edge_b, vp, ctx);
  outproj_kernel<<<1024, 256, 0, stream>>>(ctx, out_w, out_b, out);
}
